// BDHBlock_68410239091437
// MI455X (gfx1250) — matrix operand order repaired and checked by reading, not compiled, not run on hardware
//
#include <hip/hip_runtime.h>
#include <hip/hip_bf16.h>
#include <cstdint>

typedef __bf16 bf16_t;
typedef __bf16 v16bf __attribute__((ext_vector_type(16)));
typedef __bf16 v8bf  __attribute__((ext_vector_type(8)));
typedef float  v8f   __attribute__((ext_vector_type(8)));
typedef int    v4i_vs __attribute__((vector_size(4 * sizeof(int))));

#define DEV static __device__ __forceinline__

// ---------------- problem constants ----------------
constexpr int kB  = 2;
constexpr int kT  = 1024;
constexpr int kD  = 512;
constexpr int kH  = 8;
constexpr int kHD = 64;
constexpr int kNH = 4096;
constexpr int kM  = kB * kT;     // 2048 rows (b,t)
constexpr int kHN = kH * kNH;    // 32768 neurons / encoder rows

// ---------------- GEMM tiling ----------------
constexpr int BM   = 128;   // block tile M
constexpr int BN   = 64;    // block tile N
constexpr int KC   = 32;    // K chunk (one WMMA K)
constexpr int LSTR = 40;    // LDS row pitch in bf16 (80B) -> conflict-free b128 frag loads
constexpr int NT   = 256;   // 8 wave32

// ---------------- async global->LDS (gfx1250 ASYNCcnt path) ----------------
#if __has_builtin(__builtin_amdgcn_global_load_async_to_lds_b128) && \
    __has_builtin(__builtin_amdgcn_s_wait_asynccnt)
#define USE_ASYNC 1
#else
#define USE_ASYNC 0
#endif

// copy 16B (8 bf16) global -> LDS
DEV void cp128_g2l(const bf16_t* g, bf16_t* l) {
#if USE_ASYNC
  __builtin_amdgcn_global_load_async_to_lds_b128((v4i_vs*)g, (v4i_vs*)l, 0, 0);
#else
  *(uint4*)l = *(const uint4*)g;
#endif
}

DEV void wait_tiles() {
#if USE_ASYNC
  __builtin_amdgcn_s_wait_asynccnt(0);
#endif
}

// ===================== WMMA helpers =====================

DEV v8f wmma_bf(v16bf a, v16bf b, v8f c) {
  return __builtin_amdgcn_wmma_f32_16x16x32_bf16(
      /*neg_a=*/false, a, /*neg_b=*/false, b,
      /*c_mod=*/(short)0, c, /*reuse_a=*/false, /*reuse_b=*/false);
}

// Stage `rows` x 32 bf16 tile (row-major, leading dim lda) into LDS (pitch LSTR).
// 2 threads per row, 32B each, as two async b128 copies.
DEV void stage_rows(const bf16_t* __restrict__ src, long lda, bf16_t* dst,
                    int rows, int tid) {
  if (tid < rows * 2) {
    int r = tid >> 1, h = tid & 1;
    const bf16_t* s = src + (long)r * lda + h * 16;
    bf16_t* d = dst + r * LSTR + h * 16;
    cp128_g2l(s, d);
    cp128_g2l(s + 8, d + 8);
  }
}

// A fragment, 16x32 bf16 (ISA layout): lanes 0-15 row M=lane, K{0..7,16..23};
// lanes 16-31 row M=lane-16, K{8..15,24..31}.
DEV v16bf frag_a(const bf16_t* lds, int row) {
  int lane = threadIdx.x & 31;
  const bf16_t* p = lds + (row + (lane & 15)) * LSTR + ((lane < 16) ? 0 : 8);
  v8bf lo = *(const v8bf*)p;
  v8bf hi = *(const v8bf*)(p + 16);
  v16bf a;
#pragma unroll
  for (int i = 0; i < 8; i++) { a[i] = lo[i]; a[i + 8] = hi[i]; }
  return a;
}

// B fragment, 32x16 bf16: lanes 0-15 col N=lane, K{0..7,16..23}; lanes 16-31 K{8..15,24..31}.
// LDS tile is stored [n][k].
DEV v16bf frag_b(const bf16_t* lds, int col) {
  int lane = threadIdx.x & 31;
  const bf16_t* p = lds + (col + (lane & 15)) * LSTR + ((lane < 16) ? 0 : 8);
  v8bf lo = *(const v8bf*)p;
  v8bf hi = *(const v8bf*)(p + 16);
  v16bf b;
#pragma unroll
  for (int i = 0; i < 8; i++) { b[i] = lo[i]; b[i + 8] = hi[i]; }
  return b;
}

// ===================== small kernels =====================

// LayerNorm of one row of D=512, 256 threads per row. OUT = bf16 or float.
template <typename OUT>
__global__ void __launch_bounds__(NT) ln_kernel(const float* __restrict__ in,
                                                OUT* __restrict__ out) {
  const int row = blockIdx.x;
  const float* x = in + (long)row * kD;
  __shared__ float red[NT];
  const int tid = threadIdx.x;
  float v0 = x[tid], v1 = x[tid + 256];
  red[tid] = v0 + v1;
  __syncthreads();
  for (int s = 128; s > 0; s >>= 1) {
    if (tid < s) red[tid] += red[tid + s];
    __syncthreads();
  }
  float mean = red[0] * (1.0f / kD);
  __syncthreads();
  float d0 = v0 - mean, d1 = v1 - mean;
  red[tid] = d0 * d0 + d1 * d1;
  __syncthreads();
  for (int s = 128; s > 0; s >>= 1) {
    if (tid < s) red[tid] += red[tid + s];
    __syncthreads();
  }
  float r = rsqrtf(red[0] * (1.0f / kD) + 1e-5f);
  OUT* o = out + (long)row * kD;
  o[tid]       = (OUT)(d0 * r);
  o[tid + 256] = (OUT)(d1 * r);
}

__global__ void cvt_bf16_kernel(const float* __restrict__ in,
                                bf16_t* __restrict__ out, long n) {
  long i = (long)blockIdx.x * blockDim.x + threadIdx.x;
  if (i < n) out[i] = (bf16_t)in[i];
}

// Tiled transpose + f32->bf16: per z-slice (R x C f32) -> (C x R bf16),
// out[z][c][r] = in[z][r][c]. 64x64 tiles via LDS, both sides coalesced.
__global__ void __launch_bounds__(NT) transpose_cvt_kernel(
    const float* __restrict__ in, bf16_t* __restrict__ out, int R, int C) {
  __shared__ bf16_t t[64][72];
  const long z = blockIdx.z;
  const float* ib = in + z * (long)R * C;
  bf16_t* ob = out + z * (long)R * C;
  const int r0 = blockIdx.y * 64, c0 = blockIdx.x * 64;
  const int tid = threadIdx.x;
  {
    int rr = tid >> 2, cq = (tid & 3) * 16;
#pragma unroll
    for (int j = 0; j < 16; j++)
      t[rr][cq + j] = (bf16_t)ib[(long)(r0 + rr) * C + c0 + cq + j];
  }
  __syncthreads();
  {
    int cc = tid >> 2, rq = (tid & 3) * 16;
#pragma unroll
    for (int j = 0; j < 16; j++)
      ob[(long)(c0 + cc) * R + r0 + rq + j] = t[rq + j][cc];
  }
}

// RoPE: read f32 proj in (b,t, h*HD+hd) layout, write bf16 in (b,h,t,hd).
__global__ void rope_kernel(const float* __restrict__ in,
                            bf16_t* __restrict__ out) {
  long f = (long)blockIdx.x * blockDim.x + threadIdx.x;  // B*T*H*32 pairs
  int i = (int)(f & 31);
  int h = (int)((f >> 5) & 7);
  int t = (int)((f >> 8) & 1023);
  int b = (int)(f >> 18);
  long inIdx  = ((long)(b * kT + t)) * kD + h * kHD + 2 * i;
  long outIdx = (((long)(b * kH + h)) * kT + t) * kHD + 2 * i;
  float e = in[inIdx], o = in[inIdx + 1];
  // inv_freq = 10000^(-2i/64) ; ln(10000)=9.2103403719...
  float freq = __expf(-(float)(2 * i) * (9.210340371976184f / 64.0f));
  float ang = (float)t * freq;
  float s, c;
  __sincosf(ang, &s, &c);
  out[outIdx]     = (bf16_t)(e * c - o * s);
  out[outIdx + 1] = (bf16_t)(e * s + o * c);
}

// V: (b,t,h,hd) f32 -> VT (b,h,hd,t) bf16  (so P@V B-tiles are row-contiguous)
__global__ void vperm_kernel(const float* __restrict__ in,
                             bf16_t* __restrict__ out) {
  long f = (long)blockIdx.x * blockDim.x + threadIdx.x;  // B*T*H*HD
  int hd = (int)(f & 63);
  int h  = (int)((f >> 6) & 7);
  int t  = (int)((f >> 9) & 1023);
  int b  = (int)(f >> 19);
  out[(((long)(b * kH + h) * kHD) + hd) * kT + t] = (bf16_t)in[f];
}

// Row softmax over 1024 scores (already causally masked with -3e38), -> bf16 P.
__global__ void __launch_bounds__(NT) softmax_kernel(const float* __restrict__ S,
                                                     bf16_t* __restrict__ P) {
  long row = blockIdx.x;
  const float* s = S + row * (long)kT;
  bf16_t* p = P + row * (long)kT;
  const int tid = threadIdx.x;
  __shared__ float red[NT];
  float v[4];
  float mx = -3.4e38f;
#pragma unroll
  for (int j = 0; j < 4; j++) { v[j] = s[tid + j * 256]; mx = fmaxf(mx, v[j]); }
  red[tid] = mx;
  __syncthreads();
  for (int st = 128; st > 0; st >>= 1) {
    if (tid < st) red[tid] = fmaxf(red[tid], red[tid + st]);
    __syncthreads();
  }
  mx = red[0];
  __syncthreads();
  float sum = 0.f;
#pragma unroll
  for (int j = 0; j < 4; j++) { v[j] = __expf(v[j] - mx); sum += v[j]; }
  red[tid] = sum;
  __syncthreads();
  for (int st = 128; st > 0; st >>= 1) {
    if (tid < st) red[tid] += red[tid + st];
    __syncthreads();
  }
  float inv = 1.0f / red[0];
#pragma unroll
  for (int j = 0; j < 4; j++) p[tid + j * 256] = (bf16_t)(v[j] * inv);
}

// ===================== GEMM kernels (async double-buffered) =====================

// C(MxN,f32) = A(MxK bf16, ld=K) @ W^T, W is (NxK bf16, ld=K). q/k/v/o proj.
__global__ void __launch_bounds__(NT) gemm_transB_f32_kernel(
    const bf16_t* __restrict__ A, const bf16_t* __restrict__ W,
    float* __restrict__ C, int K, int N) {
  __shared__ bf16_t lA[2][BM * LSTR];
  __shared__ bf16_t lB[2][BN * LSTR];
  const int tid = threadIdx.x, lane = tid & 31, wid = tid >> 5;
  const int wm = wid >> 1, wn = wid & 1;
  const long m0 = (long)blockIdx.y * BM, n0 = (long)blockIdx.x * BN;
  const bf16_t* Ab = A + m0 * K;
  const bf16_t* Wb = W + n0 * K;
  v8f acc[2][2] = {};
  stage_rows(Ab, K, lA[0], BM, tid);
  stage_rows(Wb, K, lB[0], BN, tid);
  const int nk = K / KC;
  for (int i = 0; i < nk; i++) {
    wait_tiles();
    __syncthreads();
    if (i + 1 < nk) {
      stage_rows(Ab + (i + 1) * KC, K, lA[(i + 1) & 1], BM, tid);
      stage_rows(Wb + (i + 1) * KC, K, lB[(i + 1) & 1], BN, tid);
    }
    const bf16_t* cA = lA[i & 1];
    const bf16_t* cB = lB[i & 1];
    v16bf a0 = frag_a(cA, wm * 32), a1 = frag_a(cA, wm * 32 + 16);
    v16bf b0 = frag_b(cB, wn * 32), b1 = frag_b(cB, wn * 32 + 16);
    acc[0][0] = wmma_bf(a0, b0, acc[0][0]);
    acc[0][1] = wmma_bf(a0, b1, acc[0][1]);
    acc[1][0] = wmma_bf(a1, b0, acc[1][0]);
    acc[1][1] = wmma_bf(a1, b1, acc[1][1]);
    __syncthreads();
  }
  const int nl = lane & 15, mb = (lane < 16) ? 0 : 8;
#pragma unroll
  for (int mi = 0; mi < 2; mi++)
#pragma unroll
    for (int ni = 0; ni < 2; ni++)
#pragma unroll
      for (int r = 0; r < 8; r++) {
        long m = m0 + wm * 32 + mi * 16 + mb + r;
        long n = n0 + wn * 32 + ni * 16 + nl;
        C[m * N + n] = acc[mi][ni][r];
      }
}

// S[z] (TxT f32) = Q[z] (TxHD) @ K[z]^T with causal mask (no 1/sqrt(d) in ref).
__global__ void __launch_bounds__(NT) gemm_scores_kernel(
    const bf16_t* __restrict__ Q, const bf16_t* __restrict__ Kb,
    float* __restrict__ S) {
  const int z = blockIdx.z;
  const long m0 = (long)blockIdx.y * BM, n0 = (long)blockIdx.x * BN;
  float* Sp = S + (long)z * kT * kT;
  const int tid = threadIdx.x, lane = tid & 31, wid = tid >> 5;
  const int wm = wid >> 1, wn = wid & 1;
  const int nl = lane & 15, mb = (lane < 16) ? 0 : 8;

  if (n0 > m0 + (BM - 1)) {  // tile fully above the diagonal
#pragma unroll
    for (int mi = 0; mi < 2; mi++)
#pragma unroll
      for (int ni = 0; ni < 2; ni++)
#pragma unroll
        for (int r = 0; r < 8; r++) {
          long m = m0 + wm * 32 + mi * 16 + mb + r;
          long n = n0 + wn * 32 + ni * 16 + nl;
          Sp[m * kT + n] = -3.0e38f;
        }
    return;
  }

  __shared__ bf16_t lA[2][BM * LSTR];
  __shared__ bf16_t lB[2][BN * LSTR];
  const bf16_t* Qp = Q + (long)z * kT * kHD + m0 * kHD;
  const bf16_t* Kp = Kb + (long)z * kT * kHD + n0 * kHD;
  v8f acc[2][2] = {};
  stage_rows(Qp, kHD, lA[0], BM, tid);
  stage_rows(Kp, kHD, lB[0], BN, tid);
  const int nk = kHD / KC;  // 2
  for (int i = 0; i < nk; i++) {
    wait_tiles();
    __syncthreads();
    if (i + 1 < nk) {
      stage_rows(Qp + (i + 1) * KC, kHD, lA[(i + 1) & 1], BM, tid);
      stage_rows(Kp + (i + 1) * KC, kHD, lB[(i + 1) & 1], BN, tid);
    }
    const bf16_t* cA = lA[i & 1];
    const bf16_t* cB = lB[i & 1];
    v16bf a0 = frag_a(cA, wm * 32), a1 = frag_a(cA, wm * 32 + 16);
    v16bf b0 = frag_b(cB, wn * 32), b1 = frag_b(cB, wn * 32 + 16);
    acc[0][0] = wmma_bf(a0, b0, acc[0][0]);
    acc[0][1] = wmma_bf(a0, b1, acc[0][1]);
    acc[1][0] = wmma_bf(a1, b0, acc[1][0]);
    acc[1][1] = wmma_bf(a1, b1, acc[1][1]);
    __syncthreads();
  }
#pragma unroll
  for (int mi = 0; mi < 2; mi++)
#pragma unroll
    for (int ni = 0; ni < 2; ni++)
#pragma unroll
      for (int r = 0; r < 8; r++) {
        long m = m0 + wm * 32 + mi * 16 + mb + r;
        long n = n0 + wn * 32 + ni * 16 + nl;
        Sp[m * kT + n] = (n <= m) ? acc[mi][ni][r] : -3.0e38f;
      }
}

// ctx = P[z] (TxT bf16) @ V[z], V staged from VT (b,h,hd,t); out (b,t,h*HD+hd).
__global__ void __launch_bounds__(NT) gemm_ctx_kernel(
    const bf16_t* __restrict__ P, const bf16_t* __restrict__ VT,
    bf16_t* __restrict__ CTX) {
  const int z = blockIdx.z, b = z >> 3, h = z & 7;
  const long m0 = (long)blockIdx.y * BM;  // N = HD = 64 = BN -> grid.x == 1
  const bf16_t* Pp = P + (long)z * kT * kT + m0 * kT;
  const bf16_t* Vp = VT + (long)z * kHD * kT;   // rows hd, ld = kT
  __shared__ bf16_t lA[2][BM * LSTR];
  __shared__ bf16_t lB[2][BN * LSTR];
  const int tid = threadIdx.x, lane = tid & 31, wid = tid >> 5;
  const int wm = wid >> 1, wn = wid & 1;
  v8f acc[2][2] = {};
  int kmax = (int)m0 + BM;          // causal: P[m][k]==0 for k>m
  if (kmax > kT) kmax = kT;
  const int nk = kmax / KC;
  stage_rows(Pp, kT, lA[0], BM, tid);
  stage_rows(Vp, kT, lB[0], BN, tid);
  for (int i = 0; i < nk; i++) {
    wait_tiles();
    __syncthreads();
    if (i + 1 < nk) {
      stage_rows(Pp + (i + 1) * KC, kT, lA[(i + 1) & 1], BM, tid);
      stage_rows(Vp + (i + 1) * KC, kT, lB[(i + 1) & 1], BN, tid);
    }
    const bf16_t* cA = lA[i & 1];
    const bf16_t* cB = lB[i & 1];
    v16bf a0 = frag_a(cA, wm * 32), a1 = frag_a(cA, wm * 32 + 16);
    v16bf b0 = frag_b(cB, wn * 32), b1 = frag_b(cB, wn * 32 + 16);
    acc[0][0] = wmma_bf(a0, b0, acc[0][0]);
    acc[0][1] = wmma_bf(a0, b1, acc[0][1]);
    acc[1][0] = wmma_bf(a1, b0, acc[1][0]);
    acc[1][1] = wmma_bf(a1, b1, acc[1][1]);
    __syncthreads();
  }
  const int nl = lane & 15, mb = (lane < 16) ? 0 : 8;
#pragma unroll
  for (int mi = 0; mi < 2; mi++)
#pragma unroll
    for (int ni = 0; ni < 2; ni++)
#pragma unroll
      for (int r = 0; r < 8; r++) {
        long t = m0 + wm * 32 + mi * 16 + mb + r;
        long n = wn * 32 + ni * 16 + nl;
        CTX[((long)b * kT + t) * kD + h * kHD + n] = (bf16_t)acc[mi][ni][r];
      }
}

// z = relu(yn @ Wdy) * relu(xn @ Wdx)  -- dual-accumulator fused neuron GEMMs.
// A1,A2: (2048x512), B1t,B2t: transposed weights (32768x512), Z: (2048x32768).
__global__ void __launch_bounds__(NT) gemm_dual_kernel(
    const bf16_t* __restrict__ A1, const bf16_t* __restrict__ A2,
    const bf16_t* __restrict__ B1t, const bf16_t* __restrict__ B2t,
    bf16_t* __restrict__ Z) {
  __shared__ bf16_t lA1[2][BM * LSTR], lA2[2][BM * LSTR];
  __shared__ bf16_t lB1[2][BN * LSTR], lB2[2][BN * LSTR];
  const int tid = threadIdx.x, lane = tid & 31, wid = tid >> 5;
  const int wm = wid >> 1, wn = wid & 1;
  const long m0 = (long)blockIdx.y * BM, n0 = (long)blockIdx.x * BN;
  const bf16_t* A1b = A1 + m0 * kD;
  const bf16_t* A2b = A2 + m0 * kD;
  const bf16_t* B1b = B1t + n0 * kD;
  const bf16_t* B2b = B2t + n0 * kD;
  v8f acc1[2][2] = {}, acc2[2][2] = {};
  stage_rows(A1b, kD, lA1[0], BM, tid);
  stage_rows(A2b, kD, lA2[0], BM, tid);
  stage_rows(B1b, kD, lB1[0], BN, tid);
  stage_rows(B2b, kD, lB2[0], BN, tid);
  const int nk = kD / KC;  // 16
  for (int i = 0; i < nk; i++) {
    wait_tiles();
    __syncthreads();
    if (i + 1 < nk) {
      int nb = (i + 1) & 1, ko = (i + 1) * KC;
      stage_rows(A1b + ko, kD, lA1[nb], BM, tid);
      stage_rows(A2b + ko, kD, lA2[nb], BM, tid);
      stage_rows(B1b + ko, kD, lB1[nb], BN, tid);
      stage_rows(B2b + ko, kD, lB2[nb], BN, tid);
    }
    const int cb = i & 1;
    {
      v16bf a0 = frag_a(lA1[cb], wm * 32), a1 = frag_a(lA1[cb], wm * 32 + 16);
      v16bf b0 = frag_b(lB1[cb], wn * 32), b1 = frag_b(lB1[cb], wn * 32 + 16);
      acc1[0][0] = wmma_bf(a0, b0, acc1[0][0]);
      acc1[0][1] = wmma_bf(a0, b1, acc1[0][1]);
      acc1[1][0] = wmma_bf(a1, b0, acc1[1][0]);
      acc1[1][1] = wmma_bf(a1, b1, acc1[1][1]);
    }
    {
      v16bf a0 = frag_a(lA2[cb], wm * 32), a1 = frag_a(lA2[cb], wm * 32 + 16);
      v16bf b0 = frag_b(lB2[cb], wn * 32), b1 = frag_b(lB2[cb], wn * 32 + 16);
      acc2[0][0] = wmma_bf(a0, b0, acc2[0][0]);
      acc2[0][1] = wmma_bf(a0, b1, acc2[0][1]);
      acc2[1][0] = wmma_bf(a1, b0, acc2[1][0]);
      acc2[1][1] = wmma_bf(a1, b1, acc2[1][1]);
    }
    __syncthreads();
  }
  const int nl = lane & 15, mb = (lane < 16) ? 0 : 8;
#pragma unroll
  for (int mi = 0; mi < 2; mi++)
#pragma unroll
    for (int ni = 0; ni < 2; ni++)
#pragma unroll
      for (int r = 0; r < 8; r++) {
        long m = m0 + wm * 32 + mi * 16 + mb + r;
        long n = n0 + wn * 32 + ni * 16 + nl;
        float v = fmaxf(acc1[mi][ni][r], 0.f) * fmaxf(acc2[mi][ni][r], 0.f);
        Z[m * kHN + n] = (bf16_t)v;
      }
}

// out = x + z_perm @ encoder.  Reference's transpose(0,2,1,3).reshape scrambles
// rows: row i of z_perm (per batch) has h=i/128 and, for column j,
// t=(i%128)*8 + j/4096, n=j%4096, pulled from zprod[(b,t)][h*4096+n].
// A-tiles: permuted gather (32-wide K chunks never cross the 4096 boundary, so
// each LDS row is one contiguous 64B read). B-tiles: encT (512 x 32768).
__global__ void __launch_bounds__(NT) gemm_final_kernel(
    const bf16_t* __restrict__ Z, const bf16_t* __restrict__ Et,
    const float* __restrict__ X, float* __restrict__ OUT) {
  __shared__ bf16_t lA[2][BM * LSTR];
  __shared__ bf16_t lB[2][BN * LSTR];
  const int tid = threadIdx.x, lane = tid & 31, wid = tid >> 5;
  const int wm = wid >> 1, wn = wid & 1;
  const long i0 = (long)blockIdx.y * BM, n0 = (long)blockIdx.x * BN;
  const int bb = (int)(i0 >> 10);
  const int h  = (int)((i0 & 1023) >> 7);
  const bf16_t* Eb = Et + n0 * (long)kHN;
  const int r = tid >> 1, hf = tid & 1;
  v8f acc[2][2] = {};

  // permuted A-tile issue for K-chunk k0 into buffer buf
  auto issueA = [&](int k0, int buf) {
    int jh = k0 >> 12, nq = k0 & 4095;
    long base = ((long)(bb * kT) + r * 8 + jh) * (long)kHN + (long)h * kNH + nq + hf * 16;
    bf16_t* d = lA[buf] + r * LSTR + hf * 16;
    cp128_g2l(Z + base, d);
    cp128_g2l(Z + base + 8, d + 8);
  };

  issueA(0, 0);
  stage_rows(Eb, kHN, lB[0], BN, tid);
  const int nk = kHN / KC;  // 1024
  for (int i = 0; i < nk; i++) {
    wait_tiles();
    __syncthreads();
    if (i + 1 < nk) {
      issueA((i + 1) * KC, (i + 1) & 1);
      stage_rows(Eb + (i + 1) * KC, kHN, lB[(i + 1) & 1], BN, tid);
    }
    const bf16_t* cA = lA[i & 1];
    const bf16_t* cB = lB[i & 1];
    v16bf a0 = frag_a(cA, wm * 32), a1 = frag_a(cA, wm * 32 + 16);
    v16bf b0 = frag_b(cB, wn * 32), b1 = frag_b(cB, wn * 32 + 16);
    acc[0][0] = wmma_bf(a0, b0, acc[0][0]);
    acc[0][1] = wmma_bf(a0, b1, acc[0][1]);
    acc[1][0] = wmma_bf(a1, b0, acc[1][0]);
    acc[1][1] = wmma_bf(a1, b1, acc[1][1]);
    __syncthreads();
  }
  const int nl = lane & 15, mb = (lane < 16) ? 0 : 8;
#pragma unroll
  for (int mi = 0; mi < 2; mi++)
#pragma unroll
    for (int ni = 0; ni < 2; ni++)
#pragma unroll
      for (int rr = 0; rr < 8; rr++) {
        long m = i0 + wm * 32 + mi * 16 + mb + rr;
        long n = n0 + wn * 32 + ni * 16 + nl;
        OUT[m * kD + n] = acc[mi][ni][rr] + X[m * kD + n];
      }
}

// ===================== host launcher =====================

extern "C" void kernel_launch(void* const* d_in, const int* in_sizes, int n_in,
                              void* d_out, int out_size, void* d_ws, size_t ws_size,
                              hipStream_t stream) {
  (void)in_sizes; (void)n_in; (void)out_size; (void)ws_size;
  const float* x    = (const float*)d_in[0];
  const float* Wq   = (const float*)d_in[1];
  const float* Wk   = (const float*)d_in[2];
  const float* Wv   = (const float*)d_in[3];
  const float* Wo   = (const float*)d_in[4];
  const float* decx = (const float*)d_in[5];
  const float* decy = (const float*)d_in[6];
  const float* enc  = (const float*)d_in[7];

  char* ws = (char*)d_ws;
  size_t off = 0;
  auto alloc = [&](size_t bytes) -> char* {
    char* p = ws + off;
    off = (off + bytes + 255) & ~(size_t)255;
    return p;
  };

  const size_t MD  = (size_t)kM * kD;        // 1,048,576
  const size_t DHN = (size_t)kD * kHN;       // 16,777,216
  const size_t ST  = (size_t)16 * kT * kT;   // 16,777,216 score elems

  // persistent
  bf16_t* xnB  = (bf16_t*)alloc(MD * 2);
  bf16_t* wqB  = (bf16_t*)alloc((size_t)kD * kD * 2);
  bf16_t* wkB  = (bf16_t*)alloc((size_t)kD * kD * 2);
  bf16_t* wvB  = (bf16_t*)alloc((size_t)kD * kD * 2);
  bf16_t* woB  = (bf16_t*)alloc((size_t)kD * kD * 2);
  bf16_t* wdxT = (bf16_t*)alloc(DHN * 2);    // (32768 x 512): [h*NH+n][d]
  bf16_t* wdyT = (bf16_t*)alloc(DHN * 2);
  bf16_t* encT = (bf16_t*)alloc(DHN * 2);    // (512 x 32768): [d][k]
  bf16_t* ynB  = (bf16_t*)alloc(MD * 2);
  float*  attn = (float*)alloc(MD * 4);
  float*  oacc = (float*)alloc(MD * 4);

  // reuse region: q/k/v/S/P/ctx live only until the o-proj; Z overlaps them.
  size_t rbase = off;
  float*  qf  = (float*)alloc(MD * 4);
  float*  kf  = (float*)alloc(MD * 4);
  float*  vf  = (float*)alloc(MD * 4);
  bf16_t* qb  = (bf16_t*)alloc(MD * 2);
  bf16_t* kb  = (bf16_t*)alloc(MD * 2);
  bf16_t* vb  = (bf16_t*)alloc(MD * 2);     // VT layout (b,h,hd,t)
  float*  S   = (float*)alloc(ST * 4);
  bf16_t* P   = (bf16_t*)alloc(ST * 2);
  bf16_t* ctx = (bf16_t*)alloc(MD * 2);
  bf16_t* Zb  = (bf16_t*)(ws + rbase);       // 134 MB, reuses dead q..ctx region

  // 1) LN(x) -> bf16; weight conversions (plain + tiled transposes)
  ln_kernel<bf16_t><<<kM, NT, 0, stream>>>(x, xnB);
  cvt_bf16_kernel<<<(unsigned)((size_t)kD * kD / NT), NT, 0, stream>>>(Wq, wqB, (long)kD * kD);
  cvt_bf16_kernel<<<(unsigned)((size_t)kD * kD / NT), NT, 0, stream>>>(Wk, wkB, (long)kD * kD);
  cvt_bf16_kernel<<<(unsigned)((size_t)kD * kD / NT), NT, 0, stream>>>(Wv, wvB, (long)kD * kD);
  cvt_bf16_kernel<<<(unsigned)((size_t)kD * kD / NT), NT, 0, stream>>>(Wo, woB, (long)kD * kD);
  // decoder (H,D,NH): per-h transpose (D x NH) -> (NH x D)  => WdT[(h*NH+n)][d]
  transpose_cvt_kernel<<<dim3(kNH / 64, kD / 64, kH), NT, 0, stream>>>(decx, wdxT, kD, kNH);
  transpose_cvt_kernel<<<dim3(kNH / 64, kD / 64, kH), NT, 0, stream>>>(decy, wdyT, kD, kNH);
  // encoder (32768 x 512) -> encT (512 x 32768)
  transpose_cvt_kernel<<<dim3(kD / 64, kHN / 64, 1), NT, 0, stream>>>(enc, encT, kHN, kD);

  // 2) q/k/v projections (C = xn @ W^T), then RoPE / layout permute
  dim3 gProj(kD / BN, kM / BM);
  gemm_transB_f32_kernel<<<gProj, NT, 0, stream>>>(xnB, wqB, qf, kD, kD);
  gemm_transB_f32_kernel<<<gProj, NT, 0, stream>>>(xnB, wkB, kf, kD, kD);
  gemm_transB_f32_kernel<<<gProj, NT, 0, stream>>>(xnB, wvB, vf, kD, kD);
  rope_kernel<<<(unsigned)(MD / 2 / NT), NT, 0, stream>>>(qf, qb);
  rope_kernel<<<(unsigned)(MD / 2 / NT), NT, 0, stream>>>(kf, kb);
  vperm_kernel<<<(unsigned)(MD / NT), NT, 0, stream>>>(vf, vb);

  // 3) attention: scores (causal) -> softmax -> P@V
  dim3 gSc(kT / BN, kT / BM, kB * kH);
  gemm_scores_kernel<<<gSc, NT, 0, stream>>>(qb, kb, S);
  softmax_kernel<<<kB * kH * kT, NT, 0, stream>>>(S, P);
  dim3 gCtx(1, kT / BM, kB * kH);
  gemm_ctx_kernel<<<gCtx, NT, 0, stream>>>(P, vb, ctx);

  // 4) output projection + LN -> yn
  gemm_transB_f32_kernel<<<gProj, NT, 0, stream>>>(ctx, woB, attn, kD, kD);
  ln_kernel<bf16_t><<<kM, NT, 0, stream>>>(attn, ynB);

  // 5) fused dual neuron GEMM: z = relu(yn@Wdy) * relu(xn@Wdx)
  dim3 gDual(kHN / BN, kM / BM);
  gemm_dual_kernel<<<gDual, NT, 0, stream>>>(ynB, xnB, wdyT, wdxT, Zb);

  // 6) out = x + z_perm @ encoder, then final LN -> d_out
  dim3 gFin(kD / BN, kM / BM);
  gemm_final_kernel<<<gFin, NT, 0, stream>>>(Zb, encT, x, oacc);
  ln_kernel<float><<<kM, NT, 0, stream>>>(oacc, (float*)d_out);
}
